// DiffModule_40827959116531
// MI455X (gfx1250) — matrix operand order repaired and checked by reading, not compiled, not run on hardware
//
#include <hip/hip_runtime.h>
#include <hip/hip_bf16.h>
#include <math.h>
#include <stdint.h>

// ---------------------------------------------------------------------------
// Types
// ---------------------------------------------------------------------------
typedef __bf16 bhalf;
typedef __attribute__((ext_vector_type(16))) __bf16 v16bf;
typedef __attribute__((ext_vector_type(8)))  __bf16 v8bf;
typedef __attribute__((ext_vector_type(4)))  __bf16 v4bf;
typedef __attribute__((ext_vector_type(8)))  float  v8f;

#define BM 128
#define BN 128
#define BK 32
#define LDS_P 40   // padded LDS row pitch in bf16 elems (80 B, multiple of 16 B)
#define STAGES 3   // async pipeline depth (ring of LDS buffers)

enum GemmMode { OUT_BF16 = 0, OUT_F32 = 1, OUT_BF16_TRANS = 2, OUT_SUB_BF16 = 3 };

struct GemmParams {
  const bhalf* A;   long long a_bstride;   // A: [M][K] row-major (bf16)
  const bhalf* Bt;  long long b_bstride;   // Bt: [N][K] row-major (bf16)
  const float* bias;                       // optional bias[N]
  void* C;          long long c_bstride;
  const bhalf* aux; long long aux_bstride; // for OUT_SUB_BF16: C = bf16(aux - acc)
  int K;
  int ldc, ld_aux;
  float scale;
  int relu;
  int mode;
  int trans_rows;          // rows per batch for OUT_BF16_TRANS
  long long trans_bstride; // batch stride for OUT_BF16_TRANS output
};

// CDNA5 async memory->LDS copy (16 B per lane), tracked by ASYNCcnt.
// vdst = 32-bit LDS byte address, vaddr = 64-bit global address.
#define ASYNC_G2L_B128(ldsaddr, gaddr)                                   \
  asm volatile("global_load_async_to_lds_b128 %0, %1, off"               \
               :: "v"(ldsaddr), "v"(gaddr) : "memory")
// Async ops complete in order: waiting ASYNCcnt<=4 guarantees every stage
// older than the most recently issued one (4 ops/thread) has landed in LDS.
#define WAIT_ASYNC_LE4() asm volatile("s_wait_asynccnt 4" ::: "memory")
#define WAIT_ASYNC_LE0() asm volatile("s_wait_asynccnt 0" ::: "memory")

// ---------------------------------------------------------------------------
// bf16 WMMA GEMM: C[M][N] = op( scale * (A @ Bt^T) + bias )
//   block: 256 threads (8 waves), tile 128x128, K-step 32
//   3-stage LDS ring filled with GLOBAL_LOAD_ASYNC_TO_LDS_B128:
//     iter kk issues stage kk+2, computes on stage kk, waits asynccnt<=4
//     (stage kk+1 guaranteed complete, stage kk+2 still in flight)
//   wave tile: 32(M) x 64(N) = 2x4 v_wmma_f32_16x16x32_bf16 accumulators
// ---------------------------------------------------------------------------
__global__ __launch_bounds__(256) void gemm_bf16_wmma(GemmParams p)
{
  __shared__ __align__(16) bhalf As[STAGES][BM * LDS_P];  // 3 x 10240 B
  __shared__ __align__(16) bhalf Bs[STAGES][BN * LDS_P];  // 3 x 10240 B

  const int tid  = threadIdx.x;
  const int lane = tid & 31;
  const int wave = tid >> 5;
  const int wm = wave & 3;   // 4 waves along M, 32 rows each
  const int wn = wave >> 2;  // 2 waves along N, 64 cols each
  const int bm = blockIdx.y * BM;
  const int bn = blockIdx.x * BN;
  const int batch = blockIdx.z;
  const int K = p.K;

  const bhalf* A  = p.A  + (long long)batch * p.a_bstride;
  const bhalf* Bt = p.Bt + (long long)batch * p.b_bstride;

  v8f acc[2][4] = {};

  // Staging: 128 rows x 32 cols bf16 per operand per K-step.
  // 512 16-byte chunks per operand; each thread issues 2 async copies of each
  // (4 async ops per thread per stage).
  const int srow = tid >> 2;        // 0..63 (s=0), +64 (s=1)
  const int scol = (tid & 3) * 8;   // 0,8,16,24 (bf16 units)

  auto stage_async = [&](int buf, int k0) {
    #pragma unroll
    for (int s = 0; s < 2; ++s) {
      const int r = srow + s * 64;
      const bhalf* ga = A  + (long long)(bm + r) * K + k0 + scol;
      const bhalf* gb = Bt + (long long)(bn + r) * K + k0 + scol;
      const unsigned la = (unsigned)(uintptr_t)(&As[buf][r * LDS_P + scol]);
      const unsigned lb = (unsigned)(uintptr_t)(&Bs[buf][r * LDS_P + scol]);
      ASYNC_G2L_B128(la, ga);
      ASYNC_G2L_B128(lb, gb);
    }
  };

  const int nK = K / BK;

  // Prologue: prime two stages; ensure stage 0 is resident before compute.
  stage_async(0, 0);
  if (nK > 1) {
    stage_async(1, BK);
    WAIT_ASYNC_LE4();   // stage 0 complete, stage 1 may still be in flight
  } else {
    WAIT_ASYNC_LE0();
  }
  __syncthreads();

  // WMMA fragment addressing (ISA 16-bit A 16x32 / B 32x16 layouts):
  //  A: lane (m = lane%16, khalf = lane/16) -> K[khalf*8 .. +7] ++ K[16+khalf*8 .. +7]
  //  B: lane (n = lane%16, khalf = lane/16) -> K[khalf*8 .. +7] ++ K[16+khalf*8 .. +7] of column n
  const int frow  = lane & 15;
  const int khalf = lane >> 4;

  for (int kk = 0; kk < nK; ++kk) {
    const int cur = kk % STAGES;
    const bool more = (kk + 2 < nK);
    if (more) stage_async((kk + 2) % STAGES, (kk + 2) * BK);  // prefetch depth 2

    v16bf afrag[2], bfrag[4];
    #pragma unroll
    for (int mt = 0; mt < 2; ++mt) {
      const bhalf* base = &As[cur][(wm * 32 + mt * 16 + frow) * LDS_P];
      v8bf lo = *(const v8bf*)(base + khalf * 8);
      v8bf hi = *(const v8bf*)(base + 16 + khalf * 8);
      #pragma unroll
      for (int i = 0; i < 8; ++i) { afrag[mt][i] = lo[i]; afrag[mt][8 + i] = hi[i]; }
    }
    #pragma unroll
    for (int nt = 0; nt < 4; ++nt) {
      const bhalf* base = &Bs[cur][(wn * 64 + nt * 16 + frow) * LDS_P];
      v8bf lo = *(const v8bf*)(base + khalf * 8);
      v8bf hi = *(const v8bf*)(base + 16 + khalf * 8);
      #pragma unroll
      for (int i = 0; i < 8; ++i) { bfrag[nt][i] = lo[i]; bfrag[nt][8 + i] = hi[i]; }
    }

    #pragma unroll
    for (int mt = 0; mt < 2; ++mt)
      #pragma unroll
      for (int nt = 0; nt < 4; ++nt)
        acc[mt][nt] = __builtin_amdgcn_wmma_f32_16x16x32_bf16(
            false, afrag[mt], false, bfrag[nt], (short)0, acc[mt][nt], false, false);

    // Need stage kk+1 resident for next iteration. If a newer stage (kk+2)
    // was just issued, in-order completion lets us wait <=4; otherwise the
    // newest outstanding ops ARE stage kk+1 -> wait for all of them.
    if (more) WAIT_ASYNC_LE4(); else WAIT_ASYNC_LE0();
    __syncthreads();
  }

  // Epilogue. C 16x16 f32 layout: VGPR r, lane l -> (M = r + 8*(l/16), N = l%16)
  const int ncol = lane & 15;
  const int mhi  = (lane >> 4) * 8;
  #pragma unroll
  for (int mt = 0; mt < 2; ++mt) {
    #pragma unroll
    for (int nt = 0; nt < 4; ++nt) {
      const int nglob = bn + wn * 64 + nt * 16 + ncol;
      const float bv = p.bias ? p.bias[nglob] : 0.0f;
      #pragma unroll
      for (int r = 0; r < 8; ++r) {
        const int mglob = bm + wm * 32 + mt * 16 + mhi + r;
        float v = acc[mt][nt][r] * p.scale + bv;
        if (p.relu) v = fmaxf(v, 0.0f);
        if (p.mode == OUT_BF16) {
          ((bhalf*)p.C)[(long long)batch * p.c_bstride +
                        (long long)mglob * p.ldc + nglob] = (bhalf)v;
        } else if (p.mode == OUT_F32) {
          ((float*)p.C)[(long long)batch * p.c_bstride +
                        (long long)mglob * p.ldc + nglob] = v;
        } else if (p.mode == OUT_BF16_TRANS) {
          const int b2 = mglob / p.trans_rows;
          const int ml = mglob % p.trans_rows;
          ((bhalf*)p.C)[(long long)b2 * p.trans_bstride +
                        (long long)nglob * p.trans_rows + ml] = (bhalf)v;
        } else { // OUT_SUB_BF16: store aux - value
          const float sv = (float)p.aux[(long long)batch * p.aux_bstride +
                                        (long long)mglob * p.ld_aux + nglob];
          ((bhalf*)p.C)[(long long)batch * p.c_bstride +
                        (long long)mglob * p.ldc + nglob] = (bhalf)(sv - v);
        }
      }
    }
  }
}

// ---------------------------------------------------------------------------
// f32 -> bf16 elementwise convert (vectorized x4)
// ---------------------------------------------------------------------------
__global__ __launch_bounds__(256) void cvt_f32_to_bf16(const float* __restrict__ in,
                                                       bhalf* __restrict__ out,
                                                       long long n4)
{
  const long long i = (long long)blockIdx.x * blockDim.x + threadIdx.x;
  if (i >= n4) return;
  const float4 v = ((const float4*)in)[i];
  v4bf o;
  o[0] = (bhalf)v.x; o[1] = (bhalf)v.y; o[2] = (bhalf)v.z; o[3] = (bhalf)v.w;
  ((v4bf*)out)[i] = o;
}

// ---------------------------------------------------------------------------
// Transpose + convert: Wt[n][k] = bf16(W[k][n]);  W is [K][N] f32
// ---------------------------------------------------------------------------
__global__ __launch_bounds__(256) void transpose_cvt(const float* __restrict__ in,
                                                     bhalf* __restrict__ out,
                                                     int K, int N)
{
  __shared__ float tile[32][33];
  const int kb = blockIdx.y * 32, nb = blockIdx.x * 32;
  const int tx = threadIdx.x & 31, ty = threadIdx.x >> 5; // ty: 0..7
  #pragma unroll
  for (int r = ty; r < 32; r += 8)
    tile[r][tx] = in[(long long)(kb + r) * N + nb + tx];
  __syncthreads();
  #pragma unroll
  for (int r = ty; r < 32; r += 8)
    out[(long long)(nb + r) * K + kb + tx] = (bhalf)tile[tx][r];
}

// ---------------------------------------------------------------------------
// In-place row softmax over N=1024 (score already pre-scaled by 1/sqrt(O))
// ---------------------------------------------------------------------------
__global__ __launch_bounds__(256) void softmax_inplace_1024(bhalf* s)
{
  __shared__ float red[256];
  bhalf* row = s + (long long)blockIdx.x * 1024;
  const int tid = threadIdx.x;
  float v[4], mx = -INFINITY;
  #pragma unroll
  for (int i = 0; i < 4; ++i) { v[i] = (float)row[tid + i * 256]; mx = fmaxf(mx, v[i]); }
  red[tid] = mx; __syncthreads();
  for (int off = 128; off > 0; off >>= 1) {
    if (tid < off) red[tid] = fmaxf(red[tid], red[tid + off]);
    __syncthreads();
  }
  mx = red[0]; __syncthreads();
  float sum = 0.0f;
  #pragma unroll
  for (int i = 0; i < 4; ++i) { v[i] = __expf(v[i] - mx); sum += v[i]; }
  red[tid] = sum; __syncthreads();
  for (int off = 128; off > 0; off >>= 1) {
    if (tid < off) red[tid] += red[tid + off];
    __syncthreads();
  }
  const float inv = 1.0f / red[0];
  #pragma unroll
  for (int i = 0; i < 4; ++i) row[tid + i * 256] = (bhalf)(v[i] * inv);
}

// ---------------------------------------------------------------------------
// hcat[r][0..1024) = sv[r][:]   (left half of the concat; vectorized copy)
// ---------------------------------------------------------------------------
__global__ __launch_bounds__(128) void fill_hcat_left(const bhalf* __restrict__ sv,
                                                      bhalf* __restrict__ hcat)
{
  const long long r = blockIdx.x;
  ((uint4*)(hcat + r * 2048))[threadIdx.x] =
      ((const uint4*)(sv + r * 1024))[threadIdx.x];
}

// ---------------------------------------------------------------------------
// Host-side helper
// ---------------------------------------------------------------------------
static inline void run_gemm(hipStream_t stream,
    const bhalf* A, long long a_bs, const bhalf* Bt, long long b_bs,
    const float* bias, void* C, long long c_bs, int ldc,
    const bhalf* aux, long long aux_bs, int ld_aux,
    int M, int N, int K, int batches,
    float scale, int relu, int mode, int trans_rows, long long trans_bs)
{
  GemmParams p;
  p.A = A;   p.a_bstride = a_bs;
  p.Bt = Bt; p.b_bstride = b_bs;
  p.bias = bias;
  p.C = C;   p.c_bstride = c_bs;
  p.aux = aux; p.aux_bstride = aux_bs;
  p.K = K; p.ldc = ldc; p.ld_aux = ld_aux;
  p.scale = scale; p.relu = relu; p.mode = mode;
  p.trans_rows = trans_rows; p.trans_bstride = trans_bs;
  dim3 grid(N / BN, M / BM, batches);
  gemm_bf16_wmma<<<grid, dim3(256), 0, stream>>>(p);
}

extern "C" void kernel_launch(void* const* d_in, const int* in_sizes, int n_in,
                              void* d_out, int out_size, void* d_ws, size_t ws_size,
                              hipStream_t stream)
{
  (void)in_sizes; (void)n_in; (void)out_size; (void)ws_size;

  constexpr int Bb = 32, L = 1024, Nn = 1024, D = 1024, O = 1024;
  constexpr long long BL  = (long long)Bb * L;  // 32768 src rows
  constexpr long long BNr = (long long)Bb * Nn; // 32768 trg rows

  const float* src = (const float*)d_in[0];
  const float* trg = (const float*)d_in[1];
  const float* W1  = (const float*)d_in[2];
  const float* b1  = (const float*)d_in[3];
  const float* W2  = (const float*)d_in[4];
  const float* b2  = (const float*)d_in[5];
  const float* W3a = (const float*)d_in[6];
  const float* b3a = (const float*)d_in[7];
  const float* W3b = (const float*)d_in[8];
  const float* b3b = (const float*)d_in[9];

  // Workspace carving (256-byte aligned)
  char* wsp = (char*)d_ws;
  auto take = [&](size_t bytes) -> void* {
    void* p = wsp; wsp += (bytes + 255) & ~(size_t)255; return p;
  };
  bhalf* src_b = (bhalf*)take((size_t)BL  * D * 2);       // src bf16
  bhalf* trg_b = (bhalf*)take((size_t)BNr * D * 2);       // trg bf16
  bhalf* W1t   = (bhalf*)take((size_t)D * O * 2);         // W1^T
  bhalf* W2t   = (bhalf*)take((size_t)D * O * 2);         // W2^T
  bhalf* W3at  = (bhalf*)take((size_t)2 * O * O * 2);     // W3a^T [O][2O]
  bhalf* W3bt  = (bhalf*)take((size_t)O * O * 2);         // W3b^T
  bhalf* sv    = (bhalf*)take((size_t)BL  * O * 2);       // src_value
  bhalf* tk    = (bhalf*)take((size_t)BNr * O * 2);       // trg_key
  bhalf* tvt   = (bhalf*)take((size_t)Bb * O * Nn * 2);   // trg_value^T per batch
  bhalf* sc    = (bhalf*)take((size_t)Bb * L * Nn * 2);   // score / prob (in place)
  bhalf* hcat  = (bhalf*)take((size_t)BL * 2 * O * 2);    // [sv, sv-ctx]
  bhalf* hb    = (bhalf*)take((size_t)BL * O * 2);        // relu(fc3a)

  // 1) convert activations to bf16
  cvt_f32_to_bf16<<<dim3((unsigned)(BL * D / 4 / 256)), 256, 0, stream>>>(src, src_b, BL * D / 4);
  cvt_f32_to_bf16<<<dim3((unsigned)(BNr * D / 4 / 256)), 256, 0, stream>>>(trg, trg_b, BNr * D / 4);

  // 2) transpose + convert weights to [N][K] bf16
  transpose_cvt<<<dim3(O / 32, D / 32),      256, 0, stream>>>(W1,  W1t,  D,     O);
  transpose_cvt<<<dim3(O / 32, D / 32),      256, 0, stream>>>(W2,  W2t,  D,     O);
  transpose_cvt<<<dim3(O / 32, 2 * O / 32),  256, 0, stream>>>(W3a, W3at, 2 * O, O);
  transpose_cvt<<<dim3(O / 32, O / 32),      256, 0, stream>>>(W3b, W3bt, O,     O);

  // 3) src_value = src @ W1 + b1           -> sv  [BL][O] bf16
  run_gemm(stream, src_b, 0, W1t, 0, b1, sv, 0, O, nullptr, 0, 0,
           (int)BL, O, D, 1, 1.0f, 0, OUT_BF16, 0, 0);
  // 4) trg_key = trg @ W2 + b2             -> tk  [BNr][O] bf16 (B-operand of QK^T)
  run_gemm(stream, trg_b, 0, W2t, 0, b2, tk, 0, O, nullptr, 0, 0,
           (int)BNr, O, D, 1, 1.0f, 0, OUT_BF16, 0, 0);
  // 5) trg_value = trg @ W1 + b1, stored transposed per batch -> tvt [b][O][Nn]
  run_gemm(stream, trg_b, 0, W1t, 0, b1, tvt, 0, 0, nullptr, 0, 0,
           (int)BNr, O, D, 1, 1.0f, 0, OUT_BF16_TRANS, Nn, (long long)O * Nn);
  // 6) score = (sv @ tk^T) / sqrt(O)  (batched over b)  -> sc [b][L][Nn] bf16
  run_gemm(stream, sv, (long long)L * O, tk, (long long)Nn * O, nullptr,
           sc, (long long)L * Nn, Nn, nullptr, 0, 0,
           L, Nn, O, Bb, 0.03125f, 0, OUT_BF16, 0, 0);
  // 7) softmax rows in place
  softmax_inplace_1024<<<dim3((unsigned)(Bb * L)), 256, 0, stream>>>(sc);
  // 8) ctx = prob @ trg_value; write (sv - ctx) into right half of hcat
  run_gemm(stream, sc, (long long)L * Nn, tvt, (long long)O * Nn, nullptr,
           hcat + O, (long long)L * 2 * O, 2 * O, sv, (long long)L * O, O,
           L, O, Nn, Bb, 1.0f, 0, OUT_SUB_BF16, 0, 0);
  // 9) hcat left half = sv
  fill_hcat_left<<<dim3((unsigned)BL), 128, 0, stream>>>(sv, hcat);
  // 10) h = relu(hcat @ W3a + b3a)         -> hb [BL][O] bf16  (K = 2048)
  run_gemm(stream, hcat, 0, W3at, 0, b3a, hb, 0, O, nullptr, 0, 0,
           (int)BL, O, 2 * O, 1, 1.0f, 1, OUT_BF16, 0, 0);
  // 11) out = h @ W3b + b3b                -> d_out f32
  run_gemm(stream, hb, 0, W3bt, 0, b3b, d_out, 0, O, nullptr, 0, 0,
           (int)BL, O, O, 1, 1.0f, 0, OUT_F32, 0, 0);
}
